// ShiftedWindowAttention3d_43267500539963
// MI455X (gfx1250) — compile-verified, matrix operand order checked by reading, not run on hardware
//
#include <hip/hip_runtime.h>

typedef __attribute__((ext_vector_type(16))) __bf16 v16bf;
typedef __attribute__((ext_vector_type(8)))  __bf16 v8bf;
typedef __attribute__((ext_vector_type(8)))  float  v8f;

#define NWIN  128
#define NHEAD 8
#define NTOK  392
#define MPAD  400
#define KPAD  416
#define DH    32
#define DIM   256

// ---- LDS layout for fused qkv+attention kernel (bytes) ----
#define OFF_QS   0                            // 400*32 bf16 = 25600
#define OFF_KS   (OFF_QS  + MPAD*DH*2)        // 416*32 bf16 = 26624
#define OFF_VTS  (OFF_KS  + KPAD*DH*2)        // 32*416 bf16 = 26624
#define OFF_RPB  (OFF_VTS + DH*KPAD*2)        // 2536 f32    = 10144
#define OFF_UNI  (OFF_RPB + 2536*4)           // union region (16B aligned)
#define OFF_WST  OFF_UNI                      // 96*256 bf16 = 49152  (phase 1)
#define OFF_XST  (OFF_WST + 96*DIM*2)         // 8 * 16*256 bf16 = 65536 (phase 1)
#define OFF_PBUF OFF_UNI                      // 8 * 16*416 bf16 = 106496 (phase 2)
#define SMEM_ATTN (OFF_XST + 8*16*DIM*2)      // 203680 B  (< 320KB / WGP)

#define SMEM_PROJ (DIM*DIM*2 + 16*DIM*2)      // 139264 B

// ---- WMMA fragment helpers (layouts per CDNA5 ISA §7.12.2) ----
// A (16x32 bf16, M x K): lane L holds row M=L&15; vector elems 0..7 = K=k0..k0+7,
// elems 8..15 = K=k0+16..k0+23, k0 = (L>>4)*8.  A "Bt" array stored row-major as
// N x K serves the B (32x16, K x N) operand with the exact same access pattern.
__device__ __forceinline__ v16bf load_frag(const __bf16* base, int stride) {
  const int lane = threadIdx.x & 31;
  const __bf16* p = base + (lane & 15) * stride + ((lane >> 4) << 3);
  v8bf lo = *(const v8bf*)(p);
  v8bf hi = *(const v8bf*)(p + 16);
  return __builtin_shufflevector(lo, hi, 0,1,2,3,4,5,6,7,8,9,10,11,12,13,14,15);
}

__device__ __forceinline__ v8f wmma_bf16(v16bf a, v16bf b, v8f c) {
  return __builtin_amdgcn_wmma_f32_16x16x32_bf16(false, a, false, b, (short)0, c, false, false);
}

// shift-mask region labels (rolled frame): t dim (16, ws 8, ss 4), h/w (56, ws 7, ss 3)
__device__ __forceinline__ int reg_t (int g) { return g < 8  ? 0 : (g < 12 ? 1 : 2); }
__device__ __forceinline__ int reg_hw(int g) { return g < 49 ? 0 : (g < 53 ? 1 : 2); }

// =====================================================================
// Kernel 1: fused QKV projection + windowed attention, per (window, head)
// =====================================================================
__global__ void __launch_bounds__(256, 1)
swin3d_qkv_attn(const float* __restrict__ x, const float* __restrict__ qkv_w,
                const float* __restrict__ qkv_b, const float* __restrict__ rpb,
                __bf16* __restrict__ ao)
{
  extern __shared__ char smem[];
  __bf16* qs   = (__bf16*)(smem + OFF_QS);    // [400][32]  q (pre-scaled)
  __bf16* ks   = (__bf16*)(smem + OFF_KS);    // [416][32]  k
  __bf16* vts  = (__bf16*)(smem + OFF_VTS);   // [32][416]  v transposed
  float*  rpbh = (float*) (smem + OFF_RPB);   // per-head bias column
  __bf16* wst  = (__bf16*)(smem + OFF_WST);   // [96][256] W slice (q,k,v of head)
  __bf16* xst  = (__bf16*)(smem + OFF_XST);   // 8 wave-private [16][256] x tiles
  __bf16* pbufA= (__bf16*)(smem + OFF_PBUF);  // 8 wave-private [16][416] P tiles

  const int tid  = threadIdx.x;
  const int wave = tid >> 5, lane = tid & 31;
  const int win  = blockIdx.x >> 3, head = blockIdx.x & 7;
  const int wt = win >> 6, wh = (win >> 3) & 7, ww = win & 7;

  // zero q/k/vt (covers pad rows/cols so masked lanes never see NaN garbage)
  {
    unsigned* zp = (unsigned*)smem;
    for (int i = tid; i < (OFF_RPB >> 2); i += 256) zp[i] = 0u;
  }
  // per-head relative-position-bias column into LDS
  for (int i = tid; i < 2535; i += 256) rpbh[i] = rpb[i * NHEAD + head];
  // stage W slices (rows: seg*256 + head*32 + r), fp32 -> bf16
  for (int i = tid; i < 96 * 64; i += 256) {
    int row = i >> 6, c4 = (i & 63) << 2;
    int grow = (row >> 5) * DIM + head * DH + (row & 31);
    float4 f = *(const float4*)(qkv_w + (size_t)grow * DIM + c4);
    __bf16* d = wst + row * DIM + c4;
    d[0]=(__bf16)f.x; d[1]=(__bf16)f.y; d[2]=(__bf16)f.z; d[3]=(__bf16)f.w;
  }
  __syncthreads();

  // ---------------- Phase 1: QKV via WMMA ----------------
  const float qscale = 0.17677669529663687f;   // 32^-0.5
  __bf16* xs = xst + wave * 16 * DIM;
  const int nl  = lane & 15;
  const int mh8 = (lane >> 4) << 3;
  for (int mt = wave; mt < 25; mt += 8) {
    const int m0 = mt << 4;
    // stage 16x256 x tile with roll-gather (+shift mod), fp32 -> bf16
    for (int i = lane; i < 16 * 64; i += 32) {
      int row = i >> 6, c4 = (i & 63) << 2;
      int n = m0 + row;
      __bf16* d = xs + row * DIM + c4;
      if (n < NTOK) {
        int lt = n / 49, rm = n - lt * 49, lh = rm / 7, lw = rm - lh * 7;
        int st = (wt * 8 + lt + 4) & 15;
        int sh = wh * 7 + lh + 3; if (sh >= 56) sh -= 56;
        int sw = ww * 7 + lw + 3; if (sw >= 56) sw -= 56;
        float4 f = *(const float4*)(x + (size_t)((st * 56 + sh) * 56 + sw) * DIM + c4);
        d[0]=(__bf16)f.x; d[1]=(__bf16)f.y; d[2]=(__bf16)f.z; d[3]=(__bf16)f.w;
      } else {
        d[0]=(__bf16)0.f; d[1]=(__bf16)0.f; d[2]=(__bf16)0.f; d[3]=(__bf16)0.f;
      }
    }
    #pragma unroll
    for (int nt = 0; nt < 6; ++nt) {          // q0,q1,k0,k1,v0,v1 (16 cols each)
      v8f acc = {0.f,0.f,0.f,0.f,0.f,0.f,0.f,0.f};
      #pragma unroll
      for (int kc = 0; kc < 8; ++kc) {
        v16bf a = load_frag(xs + kc * 32, DIM);
        v16bf b = load_frag(wst + nt * 16 * DIM + kc * 32, DIM);
        acc = wmma_bf16(a, b, acc);
      }
      const int seg = nt >> 1, half = nt & 1;
      const float bn = qkv_b[seg * DIM + head * DH + half * 16 + nl];
      #pragma unroll
      for (int r = 0; r < 8; ++r) {
        float v = acc[r] + bn;
        int m = m0 + mh8 + r;
        if (seg == 0)      qs[m * DH + half * 16 + nl] = (__bf16)(v * qscale);
        else if (seg == 1) ks[m * DH + half * 16 + nl] = (__bf16)v;
        else               vts[(half * 16 + nl) * KPAD + m] = (__bf16)v;
      }
    }
  }
  __syncthreads();

  // ---------------- Phase 2: S = qk^T, softmax, O = P v ----------------
  __bf16* pb = pbufA + wave * 16 * KPAD;
  for (int mt = wave; mt < 25; mt += 8) {
    const int m0 = mt << 4;
    v16bf aq = load_frag(qs + m0 * DH, DH);    // one A-frag covers full dh=32
    v8f s[26];
    #pragma unroll
    for (int jt = 0; jt < 26; ++jt) {
      v16bf bk = load_frag(ks + jt * 16 * DH, DH);
      v8f z = {0.f,0.f,0.f,0.f,0.f,0.f,0.f,0.f};
      s[jt] = wmma_bf16(aq, bk, z);
    }
    // relative-position bias + shift mask
    int rbase[8], rlbl[8]; bool rvalid[8];
    #pragma unroll
    for (int r = 0; r < 8; ++r) {
      int m = m0 + mh8 + r;
      int lt = m / 49, rm = m - lt * 49, lh = rm / 7, lw = rm - lh * 7;
      rbase[r] = (lt + 7) * 169 + (lh + 6) * 13 + (lw + 6);
      rlbl[r]  = (reg_t(wt * 8 + lt) * 3 + reg_hw(wh * 7 + lh)) * 3 + reg_hw(ww * 7 + lw);
      rvalid[r] = (m < NTOK);
    }
    #pragma unroll
    for (int jt = 0; jt < 26; ++jt) {
      int j = jt * 16 + nl;
      if (j >= NTOK) {
        #pragma unroll
        for (int r = 0; r < 8; ++r) s[jt][r] = -1e30f;   // pad columns
      } else {
        int kt = j / 49, rm = j - kt * 49, kh = rm / 7, kw = rm - kh * 7;
        int joff = kt * 169 + kh * 13 + kw;
        int jlbl = (reg_t(wt * 8 + kt) * 3 + reg_hw(wh * 7 + kh)) * 3 + reg_hw(ww * 7 + kw);
        #pragma unroll
        for (int r = 0; r < 8; ++r) {
          if (rvalid[r]) {
            float b = rpbh[rbase[r] - joff];
            s[jt][r] += b + ((rlbl[r] == jlbl) ? 0.f : -100.f);
          }
        }
      }
    }
    // row softmax: C-frag rows live across 16 lanes (halves independent)
    float rmax[8], rinv[8];
    #pragma unroll
    for (int r = 0; r < 8; ++r) {
      float mx = -3.0e38f;
      #pragma unroll
      for (int jt = 0; jt < 26; ++jt) mx = fmaxf(mx, s[jt][r]);
      mx = fmaxf(mx, __shfl_xor(mx, 1)); mx = fmaxf(mx, __shfl_xor(mx, 2));
      mx = fmaxf(mx, __shfl_xor(mx, 4)); mx = fmaxf(mx, __shfl_xor(mx, 8));
      rmax[r] = mx;
    }
    #pragma unroll
    for (int jt = 0; jt < 26; ++jt)
      #pragma unroll
      for (int r = 0; r < 8; ++r) s[jt][r] = __expf(s[jt][r] - rmax[r]);
    #pragma unroll
    for (int r = 0; r < 8; ++r) {
      float sm = 0.f;
      #pragma unroll
      for (int jt = 0; jt < 26; ++jt) sm += s[jt][r];
      sm += __shfl_xor(sm, 1); sm += __shfl_xor(sm, 2);
      sm += __shfl_xor(sm, 4); sm += __shfl_xor(sm, 8);
      rinv[r] = 1.0f / sm;
    }
    #pragma unroll
    for (int jt = 0; jt < 26; ++jt)
      #pragma unroll
      for (int r = 0; r < 8; ++r)
        pb[(mh8 + r) * KPAD + jt * 16 + nl] = (__bf16)(s[jt][r] * rinv[r]);
    // O = P(16x416) @ V(416x32): 13 K-chunks, 2 N-tiles, B = vts rows (= d)
    v8f o0 = {0.f,0.f,0.f,0.f,0.f,0.f,0.f,0.f};
    v8f o1 = {0.f,0.f,0.f,0.f,0.f,0.f,0.f,0.f};
    #pragma unroll
    for (int kc = 0; kc < 13; ++kc) {
      v16bf ap = load_frag(pb + kc * 32, KPAD);
      v16bf b0 = load_frag(vts + kc * 32, KPAD);
      v16bf b1 = load_frag(vts + 16 * KPAD + kc * 32, KPAD);
      o0 = wmma_bf16(ap, b0, o0);
      o1 = wmma_bf16(ap, b1, o1);
    }
    #pragma unroll
    for (int r = 0; r < 8; ++r) {
      size_t base = (size_t)(win * MPAD + m0 + mh8 + r) * DIM + head * DH;
      ao[base + nl]      = (__bf16)o0[r];
      ao[base + 16 + nl] = (__bf16)o1[r];
    }
  }
}

// =====================================================================
// Kernel 2: output projection + reverse roll / window scatter
// =====================================================================
__global__ void __launch_bounds__(256)
swin3d_proj(const __bf16* __restrict__ ao, const float* __restrict__ proj_w,
            const float* __restrict__ proj_b, float* __restrict__ out)
{
  extern __shared__ char smem[];
  __bf16* pws = (__bf16*)smem;                  // [256][256] proj_w bf16
  __bf16* ast = (__bf16*)(smem + DIM * DIM * 2); // [16][256] A tile

  const int tid = threadIdx.x, wave = tid >> 5, lane = tid & 31;
  const int win = blockIdx.x >> 2, qq = blockIdx.x & 3;
  const int mt0 = (qq * 25) >> 2, mt1 = ((qq + 1) * 25) >> 2;
  const int wt = win >> 6, wh = (win >> 3) & 7, ww = win & 7;
  const int nl = lane & 15, mh8 = (lane >> 4) << 3;

  for (int i = tid; i < DIM * 64; i += 256) {
    int row = i >> 6, c4 = (i & 63) << 2;
    float4 f = *(const float4*)(proj_w + (size_t)row * DIM + c4);
    __bf16* d = pws + row * DIM + c4;
    d[0]=(__bf16)f.x; d[1]=(__bf16)f.y; d[2]=(__bf16)f.z; d[3]=(__bf16)f.w;
  }
  __syncthreads();

  for (int mt = mt0; mt < mt1; ++mt) {
    const int m0 = mt << 4;
    {
      const int4* src = (const int4*)(ao + (size_t)(win * MPAD + m0) * DIM);
      int4* dst = (int4*)ast;
      for (int i = tid; i < 512; i += 256) dst[i] = src[i];
    }
    __syncthreads();
    #pragma unroll
    for (int t2 = 0; t2 < 2; ++t2) {
      const int n0 = (wave * 2 + t2) << 4;
      v8f acc = {0.f,0.f,0.f,0.f,0.f,0.f,0.f,0.f};
      #pragma unroll
      for (int kc = 0; kc < 8; ++kc) {
        v16bf a = load_frag(ast + kc * 32, DIM);
        v16bf b = load_frag(pws + n0 * DIM + kc * 32, DIM);
        acc = wmma_bf16(a, b, acc);
      }
      const float bn = proj_b[n0 + nl];
      #pragma unroll
      for (int r = 0; r < 8; ++r) {
        int m = m0 + mh8 + r;
        if (m < NTOK) {
          int lt = m / 49, rm = m - lt * 49, lh = rm / 7, lw = rm - lh * 7;
          int t = (wt * 8 + lt + 4) & 15;
          int h = wh * 7 + lh + 3; if (h >= 56) h -= 56;
          int w = ww * 7 + lw + 3; if (w >= 56) w -= 56;
          out[(size_t)((t * 56 + h) * 56 + w) * DIM + n0 + nl] = acc[r] + bn;
        }
      }
    }
    __syncthreads();
  }
}

extern "C" void kernel_launch(void* const* d_in, const int* in_sizes, int n_in,
                              void* d_out, int out_size, void* d_ws, size_t ws_size,
                              hipStream_t stream) {
  (void)in_sizes; (void)n_in; (void)out_size; (void)ws_size;
  const float* x      = (const float*)d_in[0];
  const float* qkv_w  = (const float*)d_in[1];
  const float* qkv_b  = (const float*)d_in[2];
  const float* proj_w = (const float*)d_in[3];
  const float* proj_b = (const float*)d_in[4];
  const float* rpb    = (const float*)d_in[5];
  float*  out = (float*)d_out;
  __bf16* ao  = (__bf16*)d_ws;     // [128][400][256] bf16 = 26.2 MB scratch

  hipFuncSetAttribute((const void*)swin3d_qkv_attn,
                      hipFuncAttributeMaxDynamicSharedMemorySize, SMEM_ATTN);
  hipFuncSetAttribute((const void*)swin3d_proj,
                      hipFuncAttributeMaxDynamicSharedMemorySize, SMEM_PROJ);

  swin3d_qkv_attn<<<dim3(NWIN * NHEAD), dim3(256), SMEM_ATTN, stream>>>(
      x, qkv_w, qkv_b, rpb, ao);
  swin3d_proj<<<dim3(NWIN * 4), dim3(256), SMEM_PROJ, stream>>>(
      ao, proj_w, proj_b, out);
}
